// SwinTransformerBlock_55886114455712
// MI455X (gfx1250) — matrix operand order repaired and checked by reading, not compiled, not run on hardware
//
#include <hip/hip_runtime.h>
#include <hip/hip_bf16.h>

// ---------------------------------------------------------------------------
// Swin block for MI455X (gfx1250): all GEMMs + attention via
// v_wmma_f32_16x16x32_bf16 (wave32 WMMA), f32 accumulation.
// B (weight) panels staged to padded LDS per-workgroup with the CDNA5 async
// path (global_load_async_to_lds_b128 + s_wait_asynccnt); A loads pipelined;
// each wave computes a 32x64 strip (8 co-executable WMMAs per K-chunk).
// ---------------------------------------------------------------------------

typedef __bf16 bf16_t;
typedef __attribute__((ext_vector_type(16))) __bf16 v16bf;
typedef __attribute__((ext_vector_type(8)))  __bf16 v8bf;
typedef __attribute__((ext_vector_type(8)))  float  v8f;

// Problem constants
#define BB   8
#define CC   192
#define HH_  128
#define WW_  128
#define NHH  6
#define HDD  32
#define HIDD 768
#define NTOK 131072          // B*H*W tokens / pixels
#define NWIN 2048            // B*(H/8)*(W/8)

// ---------------------------------------------------------------------------
// Workspace layout (bytes, 256-aligned). Regions reused across phases.
// ---------------------------------------------------------------------------
#define OFF_QKVT  ((size_t)0)                 // 576*192*2   = 221184
#define OFF_PROJT ((size_t)221184)            // 192*192*2   = 73728
#define OFF_FC1T  ((size_t)294912)            // 768*192*2   = 294912
#define OFF_FC2T  ((size_t)589824)            // 192*768*2   = 294912
#define OFF_A     ((size_t)884736)            // xw then att : 131072*192*2 = 50331648
#define OFF_Q     ((size_t)(884736 + 50331648))           // q  : 50331648
#define OFF_K     ((size_t)(884736 + 2*50331648ull))      // k  : 50331648
#define OFF_V     ((size_t)(884736 + 3*50331648ull))      // vt : 50331648
#define OFF_XN2   OFF_Q                                   // reuses q after attention
#define OFF_H     OFF_K                                   // hidden 131072*768*2 (over k,vt)
#define OFF_X2    ((size_t)(884736 + 50331648 + 50331648 + 201326592ull))  // f32 x2

// ---------------------------------------------------------------------------
// WMMA operand loaders (CDNA5 16-bit layouts, wave32)
//   A 16x32: lane l -> row = l&15, k-half = l>>4
//     VGPR0..3 : K pairs {0,1..6,7} (+8 for upper half-lanes)
//     VGPR4..7 : K pairs {16..23}   (+8 for upper half-lanes)
//   B 32x16: lane l -> col = l&15, k-half = l>>4; same K order as A
// ---------------------------------------------------------------------------
__device__ __forceinline__ v16bf load_a16(const bf16_t* base, int stride) {
  const int lane = threadIdx.x & 31;
  const int r  = lane & 15;
  const int kh = lane >> 4;
  const bf16_t* p = base + r * stride + kh * 8;
  v8bf lo = *(const v8bf*)(p);        // K = kh*8 + 0..7
  v8bf hi = *(const v8bf*)(p + 16);   // K = 16 + kh*8 + 0..7
  return __builtin_shufflevector(lo, hi, 0,1,2,3,4,5,6,7,8,9,10,11,12,13,14,15);
}

__device__ __forceinline__ v16bf load_b16(const bf16_t* base, int stride) {
  const int lane = threadIdx.x & 31;
  const int n  = lane & 15;
  const int kh = lane >> 4;
  const bf16_t* p = base + n * stride + kh * 8;
  v8bf lo = *(const v8bf*)(p);        // 16B-aligned is enough (padded LDS rows)
  v8bf hi = *(const v8bf*)(p + 16);
  return __builtin_shufflevector(lo, hi, 0,1,2,3,4,5,6,7,8,9,10,11,12,13,14,15);
}

// ---------------------------------------------------------------------------
// Async stage of a 64-row x KDIM bf16 weight panel into LDS with +8 bf16 row
// padding (breaks the 64-bank conflict pattern of 384B strides). Each lane
// issues 16B global->LDS async copies (ASYNCcnt), drains, then barriers.
// ---------------------------------------------------------------------------
__device__ __forceinline__ void async_cp16(bf16_t* lds_dst, const bf16_t* gsrc) {
  const unsigned int laddr = (unsigned int)(uintptr_t)lds_dst;       // LDS byte offset
  const unsigned long long gaddr = (unsigned long long)(uintptr_t)gsrc;
  asm volatile("global_load_async_to_lds_b128 %0, %1, off"
               :: "v"(laddr), "v"(gaddr) : "memory");
}

__device__ __forceinline__ void async_wait_all() {
  asm volatile("s_wait_asynccnt 0x0" ::: "memory");
}

template<int KDIM>
__device__ __forceinline__ void stage_B(const bf16_t* __restrict__ Bpanel, bf16_t* Bs) {
  constexpr int LDB  = KDIM + 8;
  constexpr int CPR  = KDIM / 8;          // 16B chunks per row
  constexpr int ITER = (64 * CPR) / 256;  // chunks per thread
#pragma unroll
  for (int j = 0; j < ITER; ++j) {
    const int i   = j * 256 + threadIdx.x;
    const int row = i / CPR, ch = i % CPR;
    async_cp16(Bs + row * LDB + ch * 8, Bpanel + (size_t)row * KDIM + ch * 8);
  }
  async_wait_all();
  __syncthreads();
}

// K-loop: wave computes a (MT*16) x (NT*16) strip; A row-major [m][k] from
// global (pipelined one chunk ahead), B rows in padded LDS (leading dim LDB).
// All NT B tiles of a chunk are loaded into distinct regs before the MT*NT
// back-to-back WMMAs so DS waits don't gate individual WMMAs.
template<int KDIM, int MT, int NT>
__device__ __forceinline__ void gemm_tiles(const bf16_t* __restrict__ A,
                                           const bf16_t* Bs,
                                           int m0, v8f (&acc)[MT][NT]) {
  constexpr int LDB = KDIM + 8;
  v16bf a[MT];
#pragma unroll
  for (int mt = 0; mt < MT; ++mt)
    a[mt] = load_a16(A + (size_t)(m0 + mt * 16) * KDIM, KDIM);
  for (int kc = 0; kc < KDIM; kc += 32) {
    v16bf acur[MT];
#pragma unroll
    for (int mt = 0; mt < MT; ++mt) acur[mt] = a[mt];
    if (kc + 32 < KDIM) {
#pragma unroll
      for (int mt = 0; mt < MT; ++mt)        // prefetch next A chunk
        a[mt] = load_a16(A + (size_t)(m0 + mt * 16) * KDIM + kc + 32, KDIM);
    }
    v16bf b[NT];
#pragma unroll
    for (int s = 0; s < NT; ++s)
      b[s] = load_b16(Bs + (size_t)(s * 16) * LDB + kc, LDB);
#pragma unroll
    for (int mt = 0; mt < MT; ++mt)
#pragma unroll
      for (int s = 0; s < NT; ++s)
        acc[mt][s] = __builtin_amdgcn_wmma_f32_16x16x32_bf16(false, acur[mt], false, b[s],
                                                             (short)0, acc[mt][s], false, false);
  }
}

// ---------------------------------------------------------------------------
// K0: weight transpose + bf16 convert: src (K x N) f32 -> dst (N x K) bf16
// ---------------------------------------------------------------------------
__global__ void __launch_bounds__(256)
k_wT(const float* __restrict__ src, bf16_t* __restrict__ dst, int K, int N) {
  int i = blockIdx.x * 256 + threadIdx.x;
  if (i >= K * N) return;
  int n = i / K, k = i % K;
  dst[i] = (bf16_t)src[(size_t)k * N + n];
}

// ---------------------------------------------------------------------------
// K1: LN1 + cyclic shift + window partition -> xw bf16 [token][C]
// One wave per output token; lane handles 6 channels.
// ---------------------------------------------------------------------------
__global__ void __launch_bounds__(256)
k_ln1_win(const float* __restrict__ x, const float* __restrict__ g,
          const float* __restrict__ be, bf16_t* __restrict__ xw) {
  const int lane = threadIdx.x & 31;
  const int t = blockIdx.x * 8 + (threadIdx.x >> 5);
  const int win = t >> 6, tok = t & 63;
  const int b  = win >> 8;
  const int wy = (win >> 4) & 15, wx = win & 15;
  const int hh = (wy * 8 + (tok >> 3) + 4) & 127;   // shifted source pixel
  const int ww = (wx * 8 + (tok & 7) + 4) & 127;
  float vals[6];
  float s = 0.f, sq = 0.f;
#pragma unroll
  for (int j = 0; j < 6; ++j) {
    int c = lane * 6 + j;
    float v = x[(((size_t)b * CC + c) * HH_ + hh) * WW_ + ww];
    vals[j] = v; s += v; sq += v * v;
  }
#pragma unroll
  for (int m = 16; m >= 1; m >>= 1) {
    s  += __shfl_xor(s,  m, 32);
    sq += __shfl_xor(sq, m, 32);
  }
  const float mean = s * (1.0f / CC);
  const float var  = sq * (1.0f / CC) - mean * mean;
  const float rstd = rsqrtf(var + 1e-5f);
#pragma unroll
  for (int j = 0; j < 6; ++j) {
    int c = lane * 6 + j;
    xw[(size_t)t * CC + c] = (bf16_t)((vals[j] - mean) * rstd * g[c] + be[c]);
  }
}

// ---------------------------------------------------------------------------
// K2: QKV GEMM (131072x192 @ 192x576) + bias; scatter into q/k (tok-major)
// and v transposed (d-major) per (window, head). q pre-scaled by HD^-0.5.
// ---------------------------------------------------------------------------
__global__ void __launch_bounds__(256)
k_qkv(const bf16_t* __restrict__ A, const bf16_t* __restrict__ Bw,
      const float* __restrict__ bias, bf16_t* __restrict__ q,
      bf16_t* __restrict__ kk, bf16_t* __restrict__ vt) {
  __shared__ alignas(32) bf16_t Bs[64 * 200];
  const int lane = threadIdx.x & 31;
  const int m0 = blockIdx.x * 256 + (threadIdx.x >> 5) * 32;
  const int n0 = blockIdx.y * 64;
  stage_B<192>(Bw + (size_t)n0 * 192, Bs);
  v8f acc[2][4] = {};
  gemm_tiles<192, 2, 4>(A, Bs, m0, acc);
#pragma unroll
  for (int s = 0; s < 4; ++s) {
    const int n = n0 + s * 16 + (lane & 15);
    const int which = n / CC;           // 0=q 1=k 2=v (tile never straddles)
    const int cc = n % CC;
    const int head = cc >> 5, d = cc & 31;
    const float bb = bias[n];
#pragma unroll
    for (int mt = 0; mt < 2; ++mt)
#pragma unroll
      for (int r = 0; r < 8; ++r) {
        const int t = m0 + mt * 16 + r + 8 * (lane >> 4);
        const int win = t >> 6, tok = t & 63;
        const size_t wh = (size_t)win * NHH + head;
        float val = acc[mt][s][r] + bb;
        if (which == 0)      q [(wh * 64 + tok) * 32 + d] = (bf16_t)(val * 0.17677669529663687f);
        else if (which == 1) kk[(wh * 64 + tok) * 32 + d] = (bf16_t)val;
        else                 vt[(wh * 32 + d) * 64 + tok] = (bf16_t)val;
      }
  }
}

// ---------------------------------------------------------------------------
// K3: attention, one wave per (window, head), 4 waves/block.
// S = qk^T (16 WMMA, K=32=HD), +rel-pos bias, shuffle softmax,
// P->LDS bf16, O = P@V (16 WMMA, K=64).
// ---------------------------------------------------------------------------
__global__ void __launch_bounds__(128)
k_attn(const bf16_t* __restrict__ q, const bf16_t* __restrict__ kk,
       const bf16_t* __restrict__ vt, const float* __restrict__ rpb,
       bf16_t* __restrict__ att) {
  __shared__ alignas(32) bf16_t P[4][64 * 64];
  const int wib  = threadIdx.x >> 5;
  const int lane = threadIdx.x & 31;
  const int nlo  = lane & 15, kh = lane >> 4;
  const int wh   = blockIdx.x * 4 + wib;
  const int win  = wh / NHH, head = wh % NHH;
  const bf16_t* qb = q  + (size_t)wh * 64 * 32;
  const bf16_t* kb = kk + (size_t)wh * 64 * 32;
  const bf16_t* vb = vt + (size_t)wh * 32 * 64;

  v16bf qa[4], kv[4];
#pragma unroll
  for (int tm = 0; tm < 4; ++tm) qa[tm] = load_a16(qb + tm * 16 * 32, 32);
#pragma unroll
  for (int tn = 0; tn < 4; ++tn) kv[tn] = load_b16(kb + tn * 16 * 32, 32);

  v8f S[4][4];
#pragma unroll
  for (int tm = 0; tm < 4; ++tm)
#pragma unroll
    for (int tn = 0; tn < 4; ++tn) {
      v8f z = {};
      S[tm][tn] = __builtin_amdgcn_wmma_f32_16x16x32_bf16(false, qa[tm], false, kv[tn],
                                                          (short)0, z, false, false);
    }

  // bias + softmax over keys (row spread across 16 lanes x 4 tiles, 1 vgpr each)
#pragma unroll
  for (int tm = 0; tm < 4; ++tm) {
#pragma unroll
    for (int r = 0; r < 8; ++r) {
      const int mtok = tm * 16 + r + 8 * kh;
      const int my = mtok >> 3, mx = mtok & 7;
      float e[4];
#pragma unroll
      for (int tn = 0; tn < 4; ++tn) {
        const int ntok = tn * 16 + nlo;
        const int ny = ntok >> 3, nx = ntok & 7;
        const int rel = (my - ny + 7) * 15 + (mx - nx + 7);
        e[tn] = S[tm][tn][r] + rpb[rel * NHH + head];
      }
      float mv = fmaxf(fmaxf(e[0], e[1]), fmaxf(e[2], e[3]));
#pragma unroll
      for (int m = 8; m >= 1; m >>= 1) mv = fmaxf(mv, __shfl_xor(mv, m, 32));
      float sum = 0.f;
#pragma unroll
      for (int tn = 0; tn < 4; ++tn) { e[tn] = __expf(e[tn] - mv); sum += e[tn]; }
#pragma unroll
      for (int m = 8; m >= 1; m >>= 1) sum += __shfl_xor(sum, m, 32);
      const float inv = 1.f / sum;
#pragma unroll
      for (int tn = 0; tn < 4; ++tn)
        P[wib][mtok * 64 + tn * 16 + nlo] = (bf16_t)(e[tn] * inv);
    }
  }
  __syncthreads();

  // O = P @ V : out 64x32, K=64 in two 32-chunks; V stored [d][tok]
#pragma unroll
  for (int tm = 0; tm < 4; ++tm) {
#pragma unroll
    for (int dn = 0; dn < 2; ++dn) {
      v8f o = {};
#pragma unroll
      for (int kc = 0; kc < 2; ++kc) {
        v16bf a = load_a16(&P[wib][tm * 16 * 64 + kc * 32], 64);
        v16bf b = load_b16(vb + dn * 16 * 64 + kc * 32, 64);
        o = __builtin_amdgcn_wmma_f32_16x16x32_bf16(false, a, false, b,
                                                    (short)0, o, false, false);
      }
#pragma unroll
      for (int r = 0; r < 8; ++r) {
        const int t = win * 64 + tm * 16 + r + 8 * kh;
        const int c = head * 32 + dn * 16 + nlo;
        att[(size_t)t * CC + c] = (bf16_t)o[r];
      }
    }
  }
}

// ---------------------------------------------------------------------------
// K4: proj GEMM + bias + window reverse / unshift + residual -> x2 (f32 ws)
// ---------------------------------------------------------------------------
__global__ void __launch_bounds__(256)
k_proj(const bf16_t* __restrict__ A, const bf16_t* __restrict__ Bw,
       const float* __restrict__ bias, const float* __restrict__ x,
       float* __restrict__ x2) {
  __shared__ alignas(32) bf16_t Bs[64 * 200];
  const int lane = threadIdx.x & 31;
  const int m0 = blockIdx.x * 256 + (threadIdx.x >> 5) * 32;
  const int n0 = blockIdx.y * 64;
  stage_B<192>(Bw + (size_t)n0 * 192, Bs);
  v8f acc[2][4] = {};
  gemm_tiles<192, 2, 4>(A, Bs, m0, acc);
#pragma unroll
  for (int s = 0; s < 4; ++s) {
    const int n = n0 + s * 16 + (lane & 15);
    const float bb = bias[n];
#pragma unroll
    for (int mt = 0; mt < 2; ++mt)
#pragma unroll
      for (int r = 0; r < 8; ++r) {
        const int t = m0 + mt * 16 + r + 8 * (lane >> 4);
        const int win = t >> 6, tok = t & 63;
        const int b  = win >> 8;
        const int wy = (win >> 4) & 15, wx = win & 15;
        const int hh = (wy * 8 + (tok >> 3) + 4) & 127;   // unshift (+SHIFT)
        const int ww = (wx * 8 + (tok & 7) + 4) & 127;
        const size_t idx = (((size_t)b * CC + n) * HH_ + hh) * WW_ + ww;
        x2[idx] = x[idx] + acc[mt][s][r] + bb;
      }
  }
}

// ---------------------------------------------------------------------------
// K5: LN2 over x2 -> xn2 bf16 [pixel][C] (pixel-major)
// ---------------------------------------------------------------------------
__global__ void __launch_bounds__(256)
k_ln2(const float* __restrict__ x2, const float* __restrict__ g,
      const float* __restrict__ be, bf16_t* __restrict__ xn2) {
  const int lane = threadIdx.x & 31;
  const int p = blockIdx.x * 8 + (threadIdx.x >> 5);
  const int b = p >> 14;
  const int hh = (p >> 7) & 127, ww = p & 127;
  float vals[6];
  float s = 0.f, sq = 0.f;
#pragma unroll
  for (int j = 0; j < 6; ++j) {
    int c = lane * 6 + j;
    float v = x2[(((size_t)b * CC + c) * HH_ + hh) * WW_ + ww];
    vals[j] = v; s += v; sq += v * v;
  }
#pragma unroll
  for (int m = 16; m >= 1; m >>= 1) {
    s  += __shfl_xor(s,  m, 32);
    sq += __shfl_xor(sq, m, 32);
  }
  const float mean = s * (1.0f / CC);
  const float var  = sq * (1.0f / CC) - mean * mean;
  const float rstd = rsqrtf(var + 1e-5f);
#pragma unroll
  for (int j = 0; j < 6; ++j) {
    int c = lane * 6 + j;
    xn2[(size_t)p * CC + c] = (bf16_t)((vals[j] - mean) * rstd * g[c] + be[c]);
  }
}

// ---------------------------------------------------------------------------
// K6: FC1 GEMM + bias + exact GELU -> hidden bf16 [pixel][768]
// ---------------------------------------------------------------------------
__global__ void __launch_bounds__(256)
k_fc1(const bf16_t* __restrict__ A, const bf16_t* __restrict__ Bw,
      const float* __restrict__ bias, bf16_t* __restrict__ hbuf) {
  __shared__ alignas(32) bf16_t Bs[64 * 200];
  const int lane = threadIdx.x & 31;
  const int m0 = blockIdx.x * 256 + (threadIdx.x >> 5) * 32;
  const int n0 = blockIdx.y * 64;
  stage_B<192>(Bw + (size_t)n0 * 192, Bs);
  v8f acc[2][4] = {};
  gemm_tiles<192, 2, 4>(A, Bs, m0, acc);
#pragma unroll
  for (int s = 0; s < 4; ++s) {
    const int n = n0 + s * 16 + (lane & 15);
    const float bb = bias[n];
#pragma unroll
    for (int mt = 0; mt < 2; ++mt)
#pragma unroll
      for (int r = 0; r < 8; ++r) {
        const int p = m0 + mt * 16 + r + 8 * (lane >> 4);
        float v = acc[mt][s][r] + bb;
        v = 0.5f * v * (1.0f + erff(v * 0.70710678118654752f));  // exact GELU
        hbuf[(size_t)p * HIDD + n] = (bf16_t)v;
      }
  }
}

// ---------------------------------------------------------------------------
// K7: FC2 GEMM + bias + final residual -> out (B,C,H,W) f32
// ---------------------------------------------------------------------------
__global__ void __launch_bounds__(256)
k_fc2(const bf16_t* __restrict__ A, const bf16_t* __restrict__ Bw,
      const float* __restrict__ bias, const float* __restrict__ x2,
      float* __restrict__ out) {
  __shared__ alignas(32) bf16_t Bs[64 * 776];   // 97 KB padded panel
  const int lane = threadIdx.x & 31;
  const int m0 = blockIdx.x * 256 + (threadIdx.x >> 5) * 32;
  const int n0 = blockIdx.y * 64;
  stage_B<768>(Bw + (size_t)n0 * 768, Bs);
  v8f acc[2][4] = {};
  gemm_tiles<768, 2, 4>(A, Bs, m0, acc);
#pragma unroll
  for (int s = 0; s < 4; ++s) {
    const int n = n0 + s * 16 + (lane & 15);
    const float bb = bias[n];
#pragma unroll
    for (int mt = 0; mt < 2; ++mt)
#pragma unroll
      for (int r = 0; r < 8; ++r) {
        const int p = m0 + mt * 16 + r + 8 * (lane >> 4);
        const int b = p >> 14;
        const int hh = (p >> 7) & 127, ww = p & 127;
        const size_t idx = (((size_t)b * CC + n) * HH_ + hh) * WW_ + ww;
        out[idx] = x2[idx] + acc[mt][s][r] + bb;
      }
  }
}

// ---------------------------------------------------------------------------
extern "C" void kernel_launch(void* const* d_in, const int* in_sizes, int n_in,
                              void* d_out, int out_size, void* d_ws, size_t ws_size,
                              hipStream_t stream) {
  const float* x      = (const float*)d_in[0];
  const float* g1     = (const float*)d_in[1];
  const float* b1     = (const float*)d_in[2];
  const float* qkv_w  = (const float*)d_in[3];
  const float* qkv_b  = (const float*)d_in[4];
  const float* proj_w = (const float*)d_in[5];
  const float* proj_b = (const float*)d_in[6];
  const float* rpb    = (const float*)d_in[7];
  const float* g2     = (const float*)d_in[8];
  const float* b2     = (const float*)d_in[9];
  const float* fc1_w  = (const float*)d_in[10];
  const float* fc1_b  = (const float*)d_in[11];
  const float* fc2_w  = (const float*)d_in[12];
  const float* fc2_b  = (const float*)d_in[13];
  float* out = (float*)d_out;

  char* ws = (char*)d_ws;
  bf16_t* qkvT  = (bf16_t*)(ws + OFF_QKVT);
  bf16_t* projT = (bf16_t*)(ws + OFF_PROJT);
  bf16_t* fc1T  = (bf16_t*)(ws + OFF_FC1T);
  bf16_t* fc2T  = (bf16_t*)(ws + OFF_FC2T);
  bf16_t* xw    = (bf16_t*)(ws + OFF_A);
  bf16_t* att   = (bf16_t*)(ws + OFF_A);
  bf16_t* qb    = (bf16_t*)(ws + OFF_Q);
  bf16_t* kb    = (bf16_t*)(ws + OFF_K);
  bf16_t* vtb   = (bf16_t*)(ws + OFF_V);
  bf16_t* xn2   = (bf16_t*)(ws + OFF_XN2);
  bf16_t* hbuf  = (bf16_t*)(ws + OFF_H);
  float*  x2    = (float*)(ws + OFF_X2);

  // Weight transposes (bf16, [n][k])
  k_wT<<<(192 * 576 + 255) / 256, 256, 0, stream>>>(qkv_w,  qkvT, 192, 576);
  k_wT<<<(192 * 192 + 255) / 256, 256, 0, stream>>>(proj_w, projT, 192, 192);
  k_wT<<<(192 * 768 + 255) / 256, 256, 0, stream>>>(fc1_w,  fc1T, 192, 768);
  k_wT<<<(768 * 192 + 255) / 256, 256, 0, stream>>>(fc2_w,  fc2T, 768, 192);

  // LN1 + shift + window partition
  k_ln1_win<<<NTOK / 8, 256, 0, stream>>>(x, g1, b1, xw);

  // QKV GEMM (M=131072, N=576, K=192); block tile 256x64
  k_qkv<<<dim3(NTOK / 256, 576 / 64), 256, 0, stream>>>(xw, qkvT, qkv_b, qb, kb, vtb);

  // Attention: 2048 windows x 6 heads, 4 waves per block
  k_attn<<<(NWIN * NHH) / 4, 128, 0, stream>>>(qb, kb, vtb, rpb, att);

  // Proj GEMM + window reverse + residual -> x2
  k_proj<<<dim3(NTOK / 256, 192 / 64), 256, 0, stream>>>(att, projT, proj_b, x, x2);

  // LN2
  k_ln2<<<NTOK / 8, 256, 0, stream>>>(x2, g2, b2, xn2);

  // FC1 + GELU (M=131072, N=768, K=192)
  k_fc1<<<dim3(NTOK / 256, 768 / 64), 256, 0, stream>>>(xn2, fc1T, fc1_b, hbuf);

  // FC2 + final residual (M=131072, N=192, K=768)
  k_fc2<<<dim3(NTOK / 256, 192 / 64), 256, 0, stream>>>(hbuf, fc2T, fc2_b, x2, out);
}
